// DynamicLayer_15779709845997
// MI455X (gfx1250) — matrix operand order repaired and checked by reading, not compiled, not run on hardware
//
#include <hip/hip_runtime.h>
#include <hip/hip_bf16.h>

// ---- problem constants ----
#define BB   8
#define NN   96
#define DEP  300
#define KPAD 320     // DEP padded to multiple of 32 for WMMA K-steps
#define DD   768
#define HH   12
#define HD   64

typedef _Float16 v8h  __attribute__((ext_vector_type(8)));
typedef _Float16 v16h __attribute__((ext_vector_type(16)));
typedef float    v8f  __attribute__((ext_vector_type(8)));

// WT (f16 transposed/padded weights) lives at the front of d_ws
#define WT_BYTES  (3u * DD * KPAD * sizeof(_Float16))   // 1,474,560 B
// merged (f16) follows
#define MERGED_OFF ((size_t)WT_BYTES)

// ---------------------------------------------------------------------------
// Kernel A: pack Wq/Wk/Wv (fp32 row-major [DEP x DD]) into f16 WT[mat][n][k]
// with k padded to KPAD (zeros for k>=DEP).  Column n of W becomes a
// contiguous K-row, which is exactly what the WMMA B-fragment wants.
// ---------------------------------------------------------------------------
__global__ __launch_bounds__(256) void pack_w(const float* __restrict__ Wq,
                                              const float* __restrict__ Wk,
                                              const float* __restrict__ Wv,
                                              _Float16* __restrict__ WT) {
    int idx = blockIdx.x * 256 + threadIdx.x;   // over 3 * KPAD * DD, (mat,k,n)
    if (idx >= 3 * KPAD * DD) return;
    int mat = idx / (KPAD * DD);
    int rem = idx - mat * (KPAD * DD);
    int k = rem / DD;
    int n = rem - k * DD;
    const float* W = (mat == 0) ? Wq : ((mat == 1) ? Wk : Wv);
    float v = (k < DEP) ? W[k * DD + n] : 0.0f;
    WT[((size_t)mat * DD + n) * KPAD + k] = (_Float16)v;
}

// ---------------------------------------------------------------------------
// A-fragment loader (16-bit A 16x32 per ISA 7.12.2):
//   lanes 0-15  : row M=lane,    K chunks {0..7, 16..23}
//   lanes 16-31 : row M=lane-16, K chunks {8..15, 24..31}
// base points at a row-major f16 matrix in LDS with row stride `stride`.
// ---------------------------------------------------------------------------
__device__ __forceinline__ v16h load_a_frag(const _Float16* base, int stride,
                                            int row, int kbase, int hi) {
    const _Float16* p = base + row * stride + kbase + hi * 8;
    v8h lo = *(const v8h*)p;
    v8h hh = *(const v8h*)(p + 16);
    return __builtin_shufflevector(lo, hh, 0, 1, 2, 3, 4, 5, 6, 7,
                                           8, 9, 10, 11, 12, 13, 14, 15);
}

// B-fragment (16-bit B 32x16): lane = column n (lanes 0-15: K {0..7, 16..23},
// lanes 16-31: K {8..15, 24..31}), matching the A-fragment K order.
// With "column n stored as contiguous row" layouts this is two 16-byte loads.
__device__ __forceinline__ v16h load_b_frag(const _Float16* colrow, int kbase, int hi) {
    const _Float16* p = colrow + kbase + hi * 8;
    v8h lo = *(const v8h*)p;
    v8h hh = *(const v8h*)(p + 16);
    return __builtin_shufflevector(lo, hh, 0, 1, 2, 3, 4, 5, 6, 7,
                                           8, 9, 10, 11, 12, 13, 14, 15);
}

#define WMMA_F16(a, b, c) \
    __builtin_amdgcn_wmma_f32_16x16x32_f16(false, (a), false, (b), (short)0, (c), false, false)

// ---------------------------------------------------------------------------
// Kernel B: one workgroup per (b, n1, head).  128 threads = 4 waves.
//  phase 0: E[b,n1] (96x300 fp32) -> LDS f16 padded 96x320
//  phase 1: Qh, Kh (96x64, row-major LDS) and Vh^T (64x96) via WMMA, +bias
//  phase 2: scores = Qh * Kh^T (96x96) -> LDS f16
//  phase 3: row-wise mask + softmax / 8 (masked rows -> uniform 1/768)
//  phase 4: merged = attn * Vh -> global f16 merged[b,n1,i, h*64+d]
// ---------------------------------------------------------------------------
__global__ __launch_bounds__(128) void attn_kernel(const float* __restrict__ edge,
                                                   const int* __restrict__ dmask,
                                                   const _Float16* __restrict__ WT,
                                                   const float* __restrict__ bq,
                                                   const float* __restrict__ bk,
                                                   const float* __restrict__ bv,
                                                   _Float16* __restrict__ mergedWS) {
    __shared__ __align__(32) _Float16 Esh[NN * KPAD];  // 61,440 B
    __shared__ __align__(32) _Float16 Qs[NN * HD];     // 12,288 B
    __shared__ __align__(32) _Float16 Ks[NN * HD];     // 12,288 B
    __shared__ __align__(32) _Float16 VsT[HD * NN];    // 12,288 B
    __shared__ __align__(32) _Float16 As[NN * NN];     // 18,432 B  (scores/attn)

    const int h    = blockIdx.x % HH;
    const int n1   = (blockIdx.x / HH) % NN;
    const int b    = blockIdx.x / (HH * NN);
    const int tid  = threadIdx.x;
    const int wave = tid >> 5;
    const int lane = tid & 31;
    const int l15  = lane & 15;
    const int hi   = lane >> 4;

    // ---- phase 0: stage E into LDS as f16 (K padded to 320 with zeros) ----
    const float* Eg = edge + ((size_t)(b * NN + n1)) * NN * DEP;
    for (int idx = tid; idx < NN * DEP; idx += 128) {
        int r = idx / DEP;
        int k = idx - r * DEP;
        Esh[r * KPAD + k] = (_Float16)Eg[idx];
    }
    for (int idx = tid; idx < NN * (KPAD - DEP); idx += 128) {
        int r = idx / (KPAD - DEP);
        int k = DEP + idx - r * (KPAD - DEP);
        Esh[r * KPAD + k] = (_Float16)0.0f;
    }
    __syncthreads();

    // ---- phase 1: Q/K/V for this head.  72 output tiles (3 mats x 6 mt x 4 nt)
    for (int t = 0; t < 18; ++t) {
        int jj  = wave + 4 * t;
        int mat = jj / 24;
        int rem = jj - mat * 24;
        int mt  = rem >> 2;
        int nt  = rem & 3;
        const float* bias = (mat == 0) ? bq : ((mat == 1) ? bk : bv);
        float bval = bias[h * HD + nt * 16 + l15];   // bias depends on column only
        v8f acc = {bval, bval, bval, bval, bval, bval, bval, bval};
        const _Float16* wrow =
            WT + ((size_t)mat * DD + (h * HD + nt * 16 + l15)) * KPAD;
        #pragma unroll
        for (int kt = 0; kt < KPAD / 32; ++kt) {
            v16h a = load_a_frag(Esh, KPAD, mt * 16 + l15, kt * 32, hi);
            v16h bf = load_b_frag(wrow, kt * 32, hi);
            acc = WMMA_F16(a, bf, acc);
        }
        int col = nt * 16 + l15;
        if (mat == 2) {
            #pragma unroll
            for (int r = 0; r < 8; ++r)
                VsT[col * NN + mt * 16 + r + 8 * hi] = (_Float16)acc[r];
        } else {
            _Float16* dst = (mat == 0) ? Qs : Ks;
            #pragma unroll
            for (int r = 0; r < 8; ++r)
                dst[(mt * 16 + r + 8 * hi) * HD + col] = (_Float16)acc[r];
        }
    }
    __syncthreads();

    // ---- phase 2: scores = Qh * Kh^T  (36 tiles: 6 mt x 6 jt, K=64) ----
    for (int t = 0; t < 9; ++t) {
        int jj = wave + 4 * t;
        int mt = jj / 6;
        int jt = jj - mt * 6;
        v8f acc = {0.f, 0.f, 0.f, 0.f, 0.f, 0.f, 0.f, 0.f};
        #pragma unroll
        for (int kt = 0; kt < 2; ++kt) {
            v16h a  = load_a_frag(Qs, HD, mt * 16 + l15, kt * 32, hi);
            // B = Kh^T: column j of B == row j of Kh (contiguous in LDS)
            v16h bf = load_b_frag(Ks + (jt * 16 + l15) * HD, kt * 32, hi);
            acc = WMMA_F16(a, bf, acc);
        }
        #pragma unroll
        for (int r = 0; r < 8; ++r)
            As[(mt * 16 + r + 8 * hi) * NN + jt * 16 + l15] = (_Float16)acc[r];
    }
    __syncthreads();

    // ---- phase 3: mask + softmax over j, then /sqrt(HD)=/8, in place ----
    if (tid < NN) {
        int i = tid;
        int m = dmask[(b * NN + n1) * NN + i];
        if (m == 0) {
            // all scores -10000 -> uniform softmax 1/96, then /8
            _Float16 u = (_Float16)(1.0f / 768.0f);
            for (int j = 0; j < NN; ++j) As[i * NN + j] = u;
        } else {
            float mx = -1e30f;
            for (int j = 0; j < NN; ++j)
                mx = fmaxf(mx, (float)As[i * NN + j]);
            float s = 0.0f;
            for (int j = 0; j < NN; ++j)
                s += __expf((float)As[i * NN + j] - mx);
            float inv = 1.0f / (8.0f * s);
            for (int j = 0; j < NN; ++j)
                As[i * NN + j] = (_Float16)(__expf((float)As[i * NN + j] - mx) * inv);
        }
    }
    __syncthreads();

    // ---- phase 4: merged = attn * Vh  (24 tiles: 6 mt x 4 nt, K=96) ----
    const size_t basebn = ((size_t)(b * NN + n1)) * NN;
    for (int t = 0; t < 6; ++t) {
        int jj = wave + 4 * t;
        int mt = jj >> 2;
        int nt = jj & 3;
        v8f acc = {0.f, 0.f, 0.f, 0.f, 0.f, 0.f, 0.f, 0.f};
        #pragma unroll
        for (int kt = 0; kt < 3; ++kt) {
            v16h a  = load_a_frag(As, NN, mt * 16 + l15, kt * 32, hi);
            // B = Vh: column d of B == row d of Vh^T (contiguous in LDS)
            v16h bf = load_b_frag(VsT + (nt * 16 + l15) * NN, kt * 32, hi);
            acc = WMMA_F16(a, bf, acc);
        }
        int dcol = h * HD + nt * 16 + l15;
        #pragma unroll
        for (int r = 0; r < 8; ++r) {
            int i = mt * 16 + r + 8 * hi;
            mergedWS[(basebn + i) * DD + dcol] = (_Float16)acc[r];
        }
    }
}

// ---------------------------------------------------------------------------
// Kernel C: gated blend of merged and merged^T.
// One wave per (b,i,j): lane covers 24 contiguous features.
// ---------------------------------------------------------------------------
__global__ __launch_bounds__(256) void combine_kernel(const _Float16* __restrict__ mws,
                                                      const float* __restrict__ Wa,
                                                      const float* __restrict__ ba,
                                                      float* __restrict__ out) {
    const int wave = threadIdx.x >> 5;
    const int lane = threadIdx.x & 31;
    const int g = blockIdx.x * 8 + wave;             // 0 .. B*N*N-1
    const int b = g / (NN * NN);
    const int rem = g - b * NN * NN;
    const int i = rem / NN;
    const int j = rem - i * NN;

    const _Float16* mer  = mws + ((size_t)((b * NN + i) * NN + j)) * DD;
    const _Float16* mert = mws + ((size_t)((b * NN + j) * NN + i)) * DD;
    const int base = lane * 24;

    float s = 0.0f;
    #pragma unroll
    for (int t = 0; t < 24; ++t) {
        float a = (float)mer[base + t];
        float c = (float)mert[base + t];
        s += a * Wa[base + t] + c * Wa[DD + base + t];
    }
    #pragma unroll
    for (int m = 16; m >= 1; m >>= 1) s += __shfl_xor(s, m, 32);
    float alph = 1.0f / (1.0f + __expf(-(s + ba[0])));

    float* op = out + ((size_t)((b * NN + i) * NN + j)) * DD + base;
    #pragma unroll
    for (int t = 0; t < 24; ++t) {
        float a = (float)mer[base + t];
        float c = (float)mert[base + t];
        op[t] = (1.0f - alph) * a + alph * c;
    }
}

// ---------------------------------------------------------------------------
extern "C" void kernel_launch(void* const* d_in, const int* in_sizes, int n_in,
                              void* d_out, int out_size, void* d_ws, size_t ws_size,
                              hipStream_t stream) {
    const float* edge  = (const float*)d_in[0];
    const int*   dmask = (const int*)d_in[1];
    const float* Wq = (const float*)d_in[2];
    const float* bq = (const float*)d_in[3];
    const float* Wk = (const float*)d_in[4];
    const float* bk = (const float*)d_in[5];
    const float* Wv = (const float*)d_in[6];
    const float* bv = (const float*)d_in[7];
    const float* Wa = (const float*)d_in[8];
    const float* ba = (const float*)d_in[9];

    _Float16* WT  = (_Float16*)d_ws;
    _Float16* mws = (_Float16*)((char*)d_ws + MERGED_OFF);
    float* out = (float*)d_out;

    // 1) pack weights to f16 transposed/padded layout
    pack_w<<<(3 * KPAD * DD + 255) / 256, 256, 0, stream>>>(Wq, Wk, Wv, WT);

    // 2) fused QKV + attention per (b, n1, head)
    attn_kernel<<<BB * NN * HH, 128, 0, stream>>>(edge, dmask, WT, bq, bk, bv, mws);

    // 3) sigmoid-gated blend of merged / merged^T
    combine_kernel<<<(BB * NN * NN) / 8, 256, 0, stream>>>(mws, Wa, ba, out);
}
